// EdgeAggregateGlobalHead_57449482551841
// MI455X (gfx1250) — matrix operand order repaired and checked by reading, not compiled, not run on hardware
//
#include <hip/hip_runtime.h>
#include <hip/hip_bf16.h>
#include <math.h>

// ---------------------------------------------------------------------------
// EdgeAggregateGlobalHead for MI455X (gfx1250, wave32, WMMA bf16, async LDS)
//
//   0) init             : zero counters / accumulators in ws
//   1) count/scan/scatter: bucket edge ids by graph b = batch[edge_index[0][e]]
//   2) kmat             : k = lrelu(u @ Wk + bk)             (256x128 fp32)
//   3) bias             : qbias/vbias = u[b] @ W[128:,:] + b (folds the concat)
//   4) wfrag            : pre-swizzle Wq/Wv top halves into plane-split
//                         WMMA B-fragment layout (bf16)
//   5) passA            : per-graph 16-edge tiles; edges staged to LDS with
//                         global_load_async_to_lds_b128 (double-issue pipeline),
//                         q via v_wmma_f32_16x16x32_bf16 (B-frags LDS-resident),
//                         s_e = q_e . k[b] / sqrt(128); store s, atomic max
//   6) passB            : v via WMMA, w = exp(s - max), register-resident
//                         per-graph accumulation; tiny atomic flush
//   7) final            : out = lrelu((acc/denom) @ Wl + bl)
// ---------------------------------------------------------------------------

#define NE    500000
#define NB    256
#define D     128
#define SPLIT 16           // blocks per graph
#define WAVES 2            // waves per block
#define TPB   (WAVES * 32)
#define RS    132          // staged row stride in floats (pad breaks bank alignment)

typedef unsigned int u32;
typedef __bf16 bf16t;
typedef __attribute__((ext_vector_type(16))) __bf16 v16bf;
typedef __attribute__((ext_vector_type(8)))  __bf16 v8bf;
typedef __attribute__((ext_vector_type(8)))  float  v8f;

// ---- workspace layout (bytes) ----
constexpr size_t al256(size_t x) { return (x + 255) & ~(size_t)255; }
constexpr size_t OFF_S      = 0;                                   // NE floats
constexpr size_t OFF_BUCKET = al256(OFF_S + (size_t)NE * 4);       // NE ints
constexpr size_t OFF_CNT    = al256(OFF_BUCKET + (size_t)NE * 4);  // 256 u32
constexpr size_t OFF_OFFS   = al256(OFF_CNT + 256 * 4);            // 257 u32
constexpr size_t OFF_CUR    = al256(OFF_OFFS + 257 * 4);           // 256 u32
constexpr size_t OFF_MAXB   = al256(OFF_CUR + 256 * 4);            // 256 u32
constexpr size_t OFF_DENOM  = al256(OFF_MAXB + 256 * 4);           // 256 f32
constexpr size_t OFF_ACC    = al256(OFF_DENOM + 256 * 4);          // 256*128 f32
constexpr size_t OFF_KMAT   = al256(OFF_ACC + 32768 * 4);          // 256*128 f32
constexpr size_t OFF_QBIAS  = al256(OFF_KMAT + 32768 * 4);         // 256*128 f32
constexpr size_t OFF_VBIAS  = al256(OFF_QBIAS + 32768 * 4);        // 256*128 f32
constexpr size_t OFF_WQF    = al256(OFF_VBIAS + 32768 * 4);        // 16384 bf16
constexpr size_t OFF_WVF    = al256(OFF_WQF + 16384 * 2);          // 16384 bf16

__device__ __forceinline__ float lrelu(float x) { return x > 0.f ? x : 0.01f * x; }

// order-preserving float<->uint encoding for atomic max on floats
__device__ __forceinline__ u32 encf(float f) {
  u32 u = __float_as_uint(f);
  return (u & 0x80000000u) ? ~u : (u | 0x80000000u);
}
__device__ __forceinline__ float decf(u32 u) {
  return (u & 0x80000000u) ? __uint_as_float(u ^ 0x80000000u) : __uint_as_float(~u);
}

// ---------------------------------------------------------------------------
// small setup kernels
// ---------------------------------------------------------------------------
__global__ void init_kernel(u32* cnt, u32* cur, u32* maxb, float* denom, float* accg) {
  int id = blockIdx.x * blockDim.x + threadIdx.x;
  if (id < NB * D) accg[id] = 0.f;
  if (id < NB) { cnt[id] = 0u; cur[id] = 0u; maxb[id] = 0u; denom[id] = 0.f; }
}

__global__ void count_kernel(const int* ei0, const int* batch, u32* cnt) {
  int e = blockIdx.x * blockDim.x + threadIdx.x;
  if (e < NE) atomicAdd(cnt + batch[ei0[e]], 1u);
}

__global__ void scan_kernel(const u32* cnt, u32* offs) {
  if (threadIdx.x == 0) {
    u32 a = 0;
    for (int b = 0; b < NB; ++b) { offs[b] = a; a += cnt[b]; }
    offs[NB] = a;
  }
}

__global__ void scatter_kernel(const int* ei0, const int* batch, const u32* offs,
                               u32* cur, int* bucket) {
  int e = blockIdx.x * blockDim.x + threadIdx.x;
  if (e >= NE) return;
  int b = batch[ei0[e]];
  u32 p = atomicAdd(cur + b, 1u);
  bucket[offs[b] + p] = e;
}

__global__ void kmat_kernel(const float* u, const float* Wk, const float* bk, float* kmat) {
  int id = blockIdx.x * blockDim.x + threadIdx.x;
  if (id >= NB * D) return;
  int c = id & (D - 1), b = id >> 7;
  float s = bk[c];
  for (int i = 0; i < D; ++i) s += u[b * D + i] * Wk[i * D + c];
  kmat[id] = lrelu(s);
}

// per-graph folded bias: qbias[b][c] = bq[c] + u[b] @ Wq[128:,:] (same for v)
__global__ void bias_kernel(const float* u, const float* Wq, const float* bq,
                            const float* Wv, const float* bv, float* qbias, float* vbias) {
  int id = blockIdx.x * blockDim.x + threadIdx.x;
  if (id >= 2 * NB * D) return;
  int c = id & (D - 1), b = (id >> 7) & (NB - 1), mat = id >> 15;
  const float* W = mat ? Wv : Wq;
  const float* bb = mat ? bv : bq;
  float s = bb[c];
  for (int i = 0; i < D; ++i) s += u[b * D + i] * W[(D + i) * D + c];
  (mat ? vbias : qbias)[b * D + c] = s;
}

// Pre-swizzle W (top 128 rows) into plane-split WMMA bf16 B-fragment layout:
//   plane0[((nt*4+kb)*32+lane)*8 + i]      = elems 0..7  of lane's v16bf
//   plane1 (offset 8192) ... + (i-8)       = elems 8..15
// where elem i of lane corresponds to K = kb*32 + (i>>3)*16 + (lane>>4)*8 + (i&7),
// column = nt*16 + (lane&15).
__global__ void wfrag_kernel(const float* Wq, const float* Wv, bf16t* wqf, bf16t* wvf) {
  int id = blockIdx.x * blockDim.x + threadIdx.x;
  if (id >= 2 * 8 * 4 * 32 * 16) return;
  int i = id & 15, lane = (id >> 4) & 31, kb = (id >> 9) & 3, nt = (id >> 11) & 7, mat = id >> 14;
  int col = nt * 16 + (lane & 15);
  int krow = kb * 32 + (i >> 3) * 16 + (lane >> 4) * 8 + (i & 7);
  const float* W = mat ? Wv : Wq;
  size_t dst = (size_t)(i < 8 ? 0 : 8192) + ((size_t)((nt * 4 + kb) * 32 + lane)) * 8 + (i & 7);
  (mat ? wvf : wqf)[dst] = (bf16t)W[krow * D + col];
}

// ---------------------------------------------------------------------------
// device helpers for the WMMA passes
// ---------------------------------------------------------------------------
// Issue 16 async row-copies (global -> LDS, f32, 16B per lane) for one tile.
// Rows past cnt are clamped to the last valid edge (finite, masked later).
__device__ __forceinline__ void issue_stage(float* xs, const float* edges, const int* bucket,
                                            int off_b, int tbase, int cnt, int lane) {
  #pragma unroll
  for (int r = 0; r < 16; ++r) {
    int slot = tbase + r;
    int cs = slot < cnt ? slot : cnt - 1;
    int e = bucket[off_b + cs];
    const float* g = edges + (size_t)e * D + lane * 4;
    unsigned lds = (unsigned)(size_t)(xs + r * RS + lane * 4);
    asm volatile("global_load_async_to_lds_b128 %0, %1, off"
                 :: "v"(lds), "v"(g) : "memory");
  }
}

// A-fragments per ISA 16-bit 16x32 layout, converting f32 -> bf16 on the fly:
// lane<16: elems0-7 = K 0..7, elems8-15 = K 16..23 ; lane>=16: K 8..15 / 24..31
__device__ __forceinline__ void build_afrags_f32(const float* xs, int lane, v16bf af[4]) {
  int row = lane & 15, half = lane >> 4;
  const float* base = xs + row * RS;
  #pragma unroll
  for (int kb = 0; kb < 4; ++kb) {
    float4 a0 = *reinterpret_cast<const float4*>(base + kb * 32 + half * 8);
    float4 a1 = *reinterpret_cast<const float4*>(base + kb * 32 + half * 8 + 4);
    float4 b0 = *reinterpret_cast<const float4*>(base + kb * 32 + 16 + half * 8);
    float4 b1 = *reinterpret_cast<const float4*>(base + kb * 32 + 16 + half * 8 + 4);
    v16bf a;
    a[0] = (bf16t)a0.x;  a[1] = (bf16t)a0.y;  a[2]  = (bf16t)a0.z;  a[3]  = (bf16t)a0.w;
    a[4] = (bf16t)a1.x;  a[5] = (bf16t)a1.y;  a[6]  = (bf16t)a1.z;  a[7]  = (bf16t)a1.w;
    a[8] = (bf16t)b0.x;  a[9] = (bf16t)b0.y;  a[10] = (bf16t)b0.z;  a[11] = (bf16t)b0.w;
    a[12] = (bf16t)b1.x; a[13] = (bf16t)b1.y; a[14] = (bf16t)b1.z;  a[15] = (bf16t)b1.w;
    af[kb] = a;
  }
}

// One 16x16 output tile: 4 chained bf16 WMMAs over K=128, B-frags from LDS.
__device__ __forceinline__ v8f wmma_nt_lds(const v16bf af[4], const bf16t* wsh,
                                           int nt, int lane) {
  v8f acc = {};
  #pragma unroll
  for (int kb = 0; kb < 4; ++kb) {
    v8bf lo = *reinterpret_cast<const v8bf*>(wsh + ((nt * 4 + kb) * 32 + lane) * 8);
    v8bf hi = *reinterpret_cast<const v8bf*>(wsh + 8192 + ((nt * 4 + kb) * 32 + lane) * 8);
    v16bf bfr;
    #pragma unroll
    for (int i = 0; i < 8; ++i) { bfr[i] = lo[i]; bfr[i + 8] = hi[i]; }
    acc = __builtin_amdgcn_wmma_f32_16x16x32_bf16(false, af[kb], false, bfr,
                                                  (short)0, acc, false, false);
  }
  return acc;
}

// ---------------------------------------------------------------------------
// pass A: q = lrelu(edges@Wq_top + qbias[b]); s_e = q_e . k[b] / sqrt(128)
// ---------------------------------------------------------------------------
__global__ __launch_bounds__(TPB) void passA_kernel(
    const float* edges, const int* bucket, const u32* offs, const bf16t* wqf,
    const float* qbias, const float* kmat, float* sarr, u32* maxbits) {
  __shared__ float xs_all[WAVES][16 * RS];
  __shared__ bf16t wsh[16384];

  { // stage weight fragments into LDS once per block
    const int* srcw = (const int*)wqf;
    int* dstw = (int*)wsh;
    for (int i = threadIdx.x; i < 8192; i += TPB) dstw[i] = srcw[i];
  }
  __syncthreads();

  int b = blockIdx.x / SPLIT, sp = blockIdx.x % SPLIT;
  int wave = threadIdx.x >> 5, lane = threadIdx.x & 31;
  int g = sp * WAVES + wave, G = SPLIT * WAVES;
  int off_b = (int)offs[b];
  int cnt = (int)offs[b + 1] - off_b;
  int ntiles = (cnt + 15) >> 4;
  int lane16 = lane & 15, half = lane >> 4;
  float* xs = xs_all[wave];

  float kreg[8], qb[8];
  #pragma unroll
  for (int nt = 0; nt < 8; ++nt) {
    kreg[nt] = kmat[b * D + nt * 16 + lane16];
    qb[nt]   = qbias[b * D + nt * 16 + lane16];
  }

  if (g < ntiles) issue_stage(xs, edges, bucket, off_b, g * 16, cnt, lane);

  float tmax = -__builtin_inff();
  for (int t = g; t < ntiles; t += G) {
    asm volatile("s_wait_asynccnt 0x0" ::: "memory");   // tile resident in LDS
    v16bf af[4];
    build_afrags_f32(xs, lane, af);
    asm volatile("s_wait_dscnt 0x0" ::: "memory");      // frag reads drained
    if (t + G < ntiles)                                  // pipeline next tile
      issue_stage(xs, edges, bucket, off_b, (t + G) * 16, cnt, lane);

    float partial[8];
    #pragma unroll
    for (int j = 0; j < 8; ++j) partial[j] = 0.f;

    #pragma unroll
    for (int nt = 0; nt < 8; ++nt) {
      v8f acc = wmma_nt_lds(af, wsh, nt, lane);
      #pragma unroll
      for (int j = 0; j < 8; ++j) {
        float qv = lrelu(acc[j] + qb[nt]);
        partial[j] += qv * kreg[nt];     // row m = half*8+j, col = nt*16+lane16
      }
    }
    #pragma unroll
    for (int j = 0; j < 8; ++j) {
      partial[j] *= 0.08838834764831845f;               // 1/sqrt(128)
      #pragma unroll
      for (int o = 1; o < 16; o <<= 1) partial[j] += __shfl_xor(partial[j], o, 32);
    }
    if (lane16 == 0) {                                  // lanes 0 (m 0..7), 16 (m 8..15)
      #pragma unroll
      for (int j = 0; j < 8; ++j) {
        int slot = t * 16 + half * 8 + j;
        if (slot < cnt) {
          sarr[off_b + slot] = partial[j];
          tmax = fmaxf(tmax, partial[j]);
        }
      }
    }
  }
  if (lane16 == 0 && tmax > -__builtin_inff()) atomicMax(maxbits + b, encf(tmax));
}

// ---------------------------------------------------------------------------
// pass B: v = lrelu(edges@Wv_top + vbias[b]); accumulate w*v and w per graph
// ---------------------------------------------------------------------------
__global__ __launch_bounds__(TPB) void passB_kernel(
    const float* edges, const int* bucket, const u32* offs, const bf16t* wvf,
    const float* vbias, const float* sarr, const u32* maxbits,
    float* accg, float* denom) {
  __shared__ float xs_all[WAVES][16 * RS];
  __shared__ bf16t wsh[16384];

  {
    const int* srcw = (const int*)wvf;
    int* dstw = (int*)wsh;
    for (int i = threadIdx.x; i < 8192; i += TPB) dstw[i] = srcw[i];
  }
  __syncthreads();

  int b = blockIdx.x / SPLIT, sp = blockIdx.x % SPLIT;
  int wave = threadIdx.x >> 5, lane = threadIdx.x & 31;
  int g = sp * WAVES + wave, G = SPLIT * WAVES;
  int off_b = (int)offs[b];
  int cnt = (int)offs[b + 1] - off_b;
  int ntiles = (cnt + 15) >> 4;
  int lane16 = lane & 15, half = lane >> 4;
  float* xs = xs_all[wave];

  float vb[8];
  #pragma unroll
  for (int nt = 0; nt < 8; ++nt) vb[nt] = vbias[b * D + nt * 16 + lane16];
  float mxb = decf(maxbits[b]);

  float accv[8];
  #pragma unroll
  for (int nt = 0; nt < 8; ++nt) accv[nt] = 0.f;
  float dloc = 0.f;

  if (g < ntiles) issue_stage(xs, edges, bucket, off_b, g * 16, cnt, lane);

  for (int t = g; t < ntiles; t += G) {
    asm volatile("s_wait_asynccnt 0x0" ::: "memory");
    v16bf af[4];
    build_afrags_f32(xs, lane, af);
    asm volatile("s_wait_dscnt 0x0" ::: "memory");
    if (t + G < ntiles)
      issue_stage(xs, edges, bucket, off_b, (t + G) * 16, cnt, lane);

    float w[8];
    #pragma unroll
    for (int j = 0; j < 8; ++j) {
      int slot = t * 16 + half * 8 + j;        // row m this lane holds
      w[j] = (slot < cnt) ? expf(sarr[off_b + slot] - mxb) : 0.f;
    }
    if (lane16 == 0) {
      #pragma unroll
      for (int j = 0; j < 8; ++j) dloc += w[j];
    }
    #pragma unroll
    for (int nt = 0; nt < 8; ++nt) {
      v8f acc = wmma_nt_lds(af, wsh, nt, lane);
      float ts = 0.f;
      #pragma unroll
      for (int j = 0; j < 8; ++j) ts += w[j] * lrelu(acc[j] + vb[nt]);
      accv[nt] += ts;                           // column nt*16+lane16, summed over rows
    }
  }
  #pragma unroll
  for (int nt = 0; nt < 8; ++nt)
    atomicAdd(accg + b * D + nt * 16 + lane16, accv[nt]);
  if (lane16 == 0) atomicAdd(denom + b, dloc);
}

// ---------------------------------------------------------------------------
// final: out = lrelu((acc/denom) @ Wl + bl)
// ---------------------------------------------------------------------------
__global__ void final_kernel(const float* accg, const float* denom, const float* Wl,
                             const float* bl, float* out) {
  int id = blockIdx.x * blockDim.x + threadIdx.x;
  if (id >= NB * D) return;
  int dcol = id & (D - 1), b = id >> 7;
  float inv = 1.f / denom[b];
  float sum = bl[dcol];
  for (int c = 0; c < D; ++c) sum += (accg[b * D + c] * inv) * Wl[c * D + dcol];
  out[id] = lrelu(sum);
}

// ---------------------------------------------------------------------------
extern "C" void kernel_launch(void* const* d_in, const int* in_sizes, int n_in,
                              void* d_out, int out_size, void* d_ws, size_t ws_size,
                              hipStream_t stream) {
  (void)in_sizes; (void)n_in; (void)out_size; (void)ws_size;
  const float* edges = (const float*)d_in[1];
  const int*   eidx  = (const int*)d_in[2];      // (2, NE), row 0 = senders
  const float* u     = (const float*)d_in[3];
  const int*   batch = (const int*)d_in[4];
  const float* Wq = (const float*)d_in[5];  const float* bq = (const float*)d_in[6];
  const float* Wk = (const float*)d_in[7];  const float* bk = (const float*)d_in[8];
  const float* Wv = (const float*)d_in[9];  const float* bv = (const float*)d_in[10];
  const float* Wl = (const float*)d_in[11]; const float* bl = (const float*)d_in[12];
  float* out = (float*)d_out;

  char* ws = (char*)d_ws;
  float* sarr  = (float*)(ws + OFF_S);
  int*   bucket= (int*)(ws + OFF_BUCKET);
  u32*   cnt   = (u32*)(ws + OFF_CNT);
  u32*   offs  = (u32*)(ws + OFF_OFFS);
  u32*   cur   = (u32*)(ws + OFF_CUR);
  u32*   maxb  = (u32*)(ws + OFF_MAXB);
  float* denom = (float*)(ws + OFF_DENOM);
  float* accg  = (float*)(ws + OFF_ACC);
  float* kmat  = (float*)(ws + OFF_KMAT);
  float* qbias = (float*)(ws + OFF_QBIAS);
  float* vbias = (float*)(ws + OFF_VBIAS);
  bf16t* wqf   = (bf16t*)(ws + OFF_WQF);
  bf16t* wvf   = (bf16t*)(ws + OFF_WVF);

  init_kernel   <<<(NB * D + 255) / 256, 256, 0, stream>>>(cnt, cur, maxb, denom, accg);
  count_kernel  <<<(NE + 255) / 256, 256, 0, stream>>>(eidx, batch, cnt);
  scan_kernel   <<<1, 32, 0, stream>>>(cnt, offs);
  scatter_kernel<<<(NE + 255) / 256, 256, 0, stream>>>(eidx, batch, offs, cur, bucket);
  kmat_kernel   <<<(NB * D + 255) / 256, 256, 0, stream>>>(u, Wk, bk, kmat);
  bias_kernel   <<<(2 * NB * D + 255) / 256, 256, 0, stream>>>(u, Wq, bq, Wv, bv, qbias, vbias);
  wfrag_kernel  <<<(32768 + 255) / 256, 256, 0, stream>>>(Wq, Wv, wqf, wvf);
  passA_kernel  <<<NB * SPLIT, TPB, 0, stream>>>(edges, bucket, offs, wqf, qbias, kmat, sarr, maxb);
  passB_kernel  <<<NB * SPLIT, TPB, 0, stream>>>(edges, bucket, offs, wvf, vbias, sarr, maxb, accg, denom);
  final_kernel  <<<(NB * D + 255) / 256, 256, 0, stream>>>(accg, denom, Wl, bl, out);
}
